// SequenceEncoder_30056181137442
// MI455X (gfx1250) — matrix operand order repaired and checked by reading, not compiled, not run on hardware
//
#include <hip/hip_runtime.h>

// ---------------------------------------------------------------------------
// SequenceEncoder for MI455X (gfx1250, wave32):
//   Phase 1: bidirectional GRU scan, 12x v_wmma_f32_16x16x32_f16 per step
//            per direction, weights resident in VGPRs (B-layout), h state
//            transposed C->A layout through 1KB LDS each step.
//   Phase 2: output projection (2 chained WMMAs, K=64) + LayerNorm(D=16)
//            via 16-lane xor-shuffle reductions.
// ---------------------------------------------------------------------------

typedef __attribute__((ext_vector_type(16))) _Float16 v16h;
typedef __attribute__((ext_vector_type(8)))  _Float16 v8h;
typedef __attribute__((ext_vector_type(8)))  float    v8f;

constexpr int Bsz  = 1024;   // batch
constexpr int Nseq = 1024;   // sequence length
constexpr int H    = 32;     // hidden per direction
constexpr int D    = 16;     // output embed dim
constexpr int NT   = Bsz / 16;  // 64 batch tiles of M=16
constexpr int TS   = 8;      // timesteps per phase-2 wave

#define LOG2E 1.44269504088896340736f

__device__ __forceinline__ float fast_sigmoid(float x) {
  float e = __builtin_amdgcn_exp2f(-x * LOG2E);
  return __builtin_amdgcn_rcpf(1.0f + e);
}
__device__ __forceinline__ float fast_tanh(float x) {
  float e = __builtin_amdgcn_exp2f(-2.0f * x * LOG2E);
  return 2.0f * __builtin_amdgcn_rcpf(1.0f + e) - 1.0f;
}
__device__ __forceinline__ v8f wmma16x16x32(v16h a, v16h b, v8f c) {
  // D = A(16x32 f16) * B(32x16 f16) + C(16x16 f32)
  return __builtin_amdgcn_wmma_f32_16x16x32_f16(false, a, false, b,
                                                (short)0, c, false, false);
}

// ---------------------------------------------------------------------------
// Phase 1: GRU scan. grid = (64 batch tiles, 2 directions), block = 1 wave32.
// Workspace layout (f16): hws[dir][bt][t][m(0..15)][hid(0..31)]  (A-friendly)
// ---------------------------------------------------------------------------
__global__ __launch_bounds__(32)
void seqenc_gru_scan(const float* __restrict__ z,
                     const float* __restrict__ Wi,    const float* __restrict__ bi,
                     const float* __restrict__ Wih_f, const float* __restrict__ Whh_f,
                     const float* __restrict__ bih_f, const float* __restrict__ bhh_f,
                     const float* __restrict__ Wih_b, const float* __restrict__ Whh_b,
                     const float* __restrict__ bih_b, const float* __restrict__ bhh_b,
                     _Float16* __restrict__ hws)
{
  const int bt   = blockIdx.x;
  const int dir  = blockIdx.y;           // 0 = forward, 1 = backward
  const int lane = threadIdx.x;          // wave32
  const int g    = lane >> 4;            // lane group (0/1)
  const int n16  = lane & 15;

  const float* Wih = dir ? Wih_b : Wih_f;
  const float* Whh = dir ? Whh_b : Whh_f;
  const float* bih = dir ? bih_b : bih_f;
  const float* bhh = dir ? bhh_b : bhh_f;

  // Preload weight tiles in WMMA B-matrix layout: lane holds col n=16j+n16,
  // element e<8 -> k = 8*g + e, e>=8 -> k = 16+8*g+(e-8). Six 16-wide gate tiles per matrix.
  v16h WihB[6], WhhB[6];
#pragma unroll
  for (int j = 0; j < 6; ++j) {
#pragma unroll
    for (int e = 0; e < 16; ++e) {
      const int k = (e < 8) ? (8 * g + e) : (16 + 8 * g + (e - 8));
      WihB[j][e] = (_Float16)Wih[(16 * j + n16) * H + k];
      WhhB[j][e] = (_Float16)Whh[(16 * j + n16) * H + k];
    }
  }

  // Biases folded into WMMA C accumulators (C layout: col = n16).
  float biasRZ[4];                       // r,z gates: b_ih + b_hh combined
#pragma unroll
  for (int j = 0; j < 4; ++j) biasRZ[j] = bih[16 * j + n16] + bhh[16 * j + n16];
  const float biasIN[2] = { bih[64 + n16], bih[80 + n16] };  // n gate, input side
  const float biasHN[2] = { bhh[64 + n16], bhh[80 + n16] };  // n gate, hidden side

  // Input-projection constants in A-layout element order:
  // e<8 -> k = 8g+e ; e>=8 -> k = 16+8g+(e-8)
  float wiA[16], biA[16];
#pragma unroll
  for (int e = 0; e < 16; ++e) {
    const int k = (e < 8) ? (8 * g + e) : (16 + 8 * g + (e - 8));
    wiA[e] = Wi[k];
    biA[e] = bi[k];
  }

  const float* zrow = z + (size_t)(bt * 16 + n16) * Nseq;

  // Hidden state: C layout (f32, two 16-wide halves) + A layout (f16).
  v8f h_c0, h_c1;
  v16h h_a;
#pragma unroll
  for (int v = 0; v < 8; ++v) { h_c0[v] = 0.0f; h_c1[v] = 0.0f; }
#pragma unroll
  for (int e = 0; e < 16; ++e) h_a[e] = (_Float16)0.0f;

  __shared__ _Float16 lds_h[16][32];     // C->A transpose staging (1 KB)

  for (int s = 0; s < Nseq; ++s) {
    const int t  = dir ? (Nseq - 1 - s) : s;
    const float zt = zrow[t];            // L2-resident after first touch

    // x_t = relu(z_t * Wi + bi), built directly in A layout (row m = n16).
    v16h xa;
#pragma unroll
    for (int e = 0; e < 16; ++e) {
      const float xv = fmaf(zt, wiA[e], biA[e]);
      xa[e] = (_Float16)(xv > 0.0f ? xv : 0.0f);
    }

    v8f hnew0, hnew1;
#pragma unroll
    for (int p = 0; p < 2; ++p) {        // hidden halves [0,16) and [16,32)
      v8f cr, cz, cin, chn;
#pragma unroll
      for (int v = 0; v < 8; ++v) {
        cr[v]  = biasRZ[p];
        cz[v]  = biasRZ[2 + p];
        cin[v] = biasIN[p];
        chn[v] = biasHN[p];
      }
      // 6 WMMAs per half: r,z chained (input+hidden), n kept split.
      v8f r  = wmma16x16x32(h_a, WhhB[p],     wmma16x16x32(xa, WihB[p],     cr));
      v8f zg = wmma16x16x32(h_a, WhhB[2 + p], wmma16x16x32(xa, WihB[2 + p], cz));
      v8f in = wmma16x16x32(xa,  WihB[4 + p], cin);
      v8f hn = wmma16x16x32(h_a, WhhB[4 + p], chn);

      const v8f hp = p ? h_c1 : h_c0;
      v8f out;
#pragma unroll
      for (int v = 0; v < 8; ++v) {
        const float rr = fast_sigmoid(r[v]);
        const float zz = fast_sigmoid(zg[v]);
        const float nn = fast_tanh(in[v] + rr * hn[v]);
        out[v] = (1.0f - zz) * nn + zz * hp[v];
      }
      if (p == 0) hnew0 = out; else hnew1 = out;
    }
    h_c0 = hnew0; h_c1 = hnew1;

    // C layout -> A layout transpose through LDS (single wave; DS in-order).
#pragma unroll
    for (int v = 0; v < 8; ++v) {
      const int m = v + 8 * g;           // C-layout row
      lds_h[m][n16]      = (_Float16)h_c0[v];
      lds_h[m][16 + n16] = (_Float16)h_c1[v];
    }
    __syncthreads();
    const v8h lo = *(const v8h*)&lds_h[n16][8 * g];        // k = 8g..8g+7
    const v8h hi = *(const v8h*)&lds_h[n16][16 + 8 * g];   // k = 16+8g..+7
    h_a = __builtin_shufflevector(lo, hi, 0,1,2,3,4,5,6,7,8,9,10,11,12,13,14,15);
    __syncthreads();

    // Persist h_t as f16 in A-friendly layout: 2x b128 stores per lane.
    _Float16* wrow = hws + ((((size_t)dir * NT + bt) * Nseq + t) * 16 + n16) * 32;
    *(v8h*)(wrow + 8 * g)      = lo;
    *(v8h*)(wrow + 16 + 8 * g) = hi;
  }
}

// ---------------------------------------------------------------------------
// Phase 2: y = [hf|hb] @ Wo^T + bo, then LayerNorm over D=16.
// grid = (64 batch tiles, 1024/TS t-chunks), block = 1 wave32.
// ---------------------------------------------------------------------------
__global__ __launch_bounds__(32)
void seqenc_proj_ln(const _Float16* __restrict__ hws,
                    const float* __restrict__ Wo, const float* __restrict__ bo,
                    const float* __restrict__ gamma, const float* __restrict__ beta,
                    float* __restrict__ out)
{
  const int bt   = blockIdx.x;
  const int tc   = blockIdx.y;
  const int lane = threadIdx.x;
  const int g    = lane >> 4;
  const int n16  = lane & 15;

  // Wo (16x64) split into two B-tiles: K 0..31 = forward h, 32..63 = backward.
  v16h Bf, Bb;
#pragma unroll
  for (int e = 0; e < 16; ++e) {
    const int k = (e < 8) ? (8 * g + e) : (16 + 8 * g + (e - 8));
    Bf[e] = (_Float16)Wo[n16 * (2 * H) + k];
    Bb[e] = (_Float16)Wo[n16 * (2 * H) + H + k];
  }
  const float gam = gamma[n16], bet = beta[n16], bon = bo[n16];
  const size_t dirStride = (size_t)NT * Nseq * 16 * 32;

  for (int tt = 0; tt < TS; ++tt) {
    const int t = tc * TS + tt;
    const _Float16* hf = hws + (((size_t)bt * Nseq + t) * 16 + n16) * 32;
    const _Float16* hb = hf + dirStride;

    const v8h flo = *(const v8h*)(hf + 8 * g);
    const v8h fhi = *(const v8h*)(hf + 16 + 8 * g);
    const v8h blo = *(const v8h*)(hb + 8 * g);
    const v8h bhi = *(const v8h*)(hb + 16 + 8 * g);
    const v16h hfA = __builtin_shufflevector(flo, fhi, 0,1,2,3,4,5,6,7,8,9,10,11,12,13,14,15);
    const v16h hbA = __builtin_shufflevector(blo, bhi, 0,1,2,3,4,5,6,7,8,9,10,11,12,13,14,15);

    v8f acc;
#pragma unroll
    for (int v = 0; v < 8; ++v) acc[v] = bon;
    acc = wmma16x16x32(hbA, Bb, acc);    // K = 32..63 (backward half)
    acc = wmma16x16x32(hfA, Bf, acc);    // K =  0..31 (forward half)

    // LayerNorm over D=16: columns live across the 16 lanes of each group.
#pragma unroll
    for (int v = 0; v < 8; ++v) {
      const float y = acc[v];
      float s = y, s2 = y * y;
#pragma unroll
      for (int mk = 1; mk < 16; mk <<= 1) {
        s  += __shfl_xor(s,  mk, 32);
        s2 += __shfl_xor(s2, mk, 32);
      }
      const float mu  = s * (1.0f / 16.0f);
      const float var = s2 * (1.0f / 16.0f) - mu * mu;
      const float inv = __builtin_amdgcn_rsqf(var + 1e-5f);
      const int m = v + 8 * g;
      out[(((size_t)(bt * 16 + m)) * Nseq + t) * D + n16] =
          (y - mu) * inv * gam + bet;
    }
  }
}

// ---------------------------------------------------------------------------
extern "C" void kernel_launch(void* const* d_in, const int* in_sizes, int n_in,
                              void* d_out, int out_size, void* d_ws, size_t ws_size,
                              hipStream_t stream) {
  const float* z     = (const float*)d_in[0];
  const float* Wi    = (const float*)d_in[1];
  const float* bi    = (const float*)d_in[2];
  const float* Wih_f = (const float*)d_in[3];
  const float* Whh_f = (const float*)d_in[4];
  const float* bih_f = (const float*)d_in[5];
  const float* bhh_f = (const float*)d_in[6];
  const float* Wih_b = (const float*)d_in[7];
  const float* Whh_b = (const float*)d_in[8];
  const float* bih_b = (const float*)d_in[9];
  const float* bhh_b = (const float*)d_in[10];
  const float* Wo    = (const float*)d_in[11];
  const float* bo    = (const float*)d_in[12];
  const float* gamma = (const float*)d_in[13];
  const float* beta  = (const float*)d_in[14];

  _Float16* hws = (_Float16*)d_ws;   // 2 * 64 * 1024 * 16 * 32 f16 = 128 MiB
  float*    out = (float*)d_out;

  seqenc_gru_scan<<<dim3(NT, 2), 32, 0, stream>>>(
      z, Wi, bi, Wih_f, Whh_f, bih_f, bhh_f,
      Wih_b, Whh_b, bih_b, bhh_b, hws);

  seqenc_proj_ln<<<dim3(NT, Nseq / TS), 32, 0, stream>>>(
      hws, Wo, bo, gamma, beta, out);
}
